// MyConv2D_1159641169954
// MI455X (gfx1250) — matrix operand order repaired and checked by reading, not compiled, not run on hardware
//
#include <hip/hip_runtime.h>

typedef __attribute__((ext_vector_type(16))) _Float16 v16h;
typedef __attribute__((ext_vector_type(8)))  _Float16 v8h;
typedef __attribute__((ext_vector_type(4)))  _Float16 v4h;
typedef __attribute__((ext_vector_type(8)))  float    v8f;

#define NB    32
#define HH    56
#define WW    56
#define CC    128
#define OHH   54
#define OWW   54
#define FF    256
#define PIX   (OHH*OWW)         /* 2916 */
#define MTOT  (NB*PIX)          /* 93312 = 1458*64 */
#define KTOT  1152
#define KSTEPS 36
#define XTOT  (NB*HH*WW*CC)     /* 12,845,056 */
#define XQ_BYTES ((size_t)XTOT * 2)
#define WQ_BYTES ((size_t)KTOT * FF * 2)
#define WS_NEEDED (XQ_BYTES + WQ_BYTES)

#if defined(__has_builtin)
#  if __has_builtin(__builtin_amdgcn_global_load_async_to_lds_b128)
#    define HAVE_ASYNC 1
#  endif
#endif
#ifndef HAVE_ASYNC
#  define HAVE_ASYNC 0
#endif

// Fixed-point quantizer: clip(round(v*256)/256, -1, 1). rintf == RNE == jnp.round.
__device__ __forceinline__ float qf(float v) {
    float r = rintf(v * 256.0f) * 0.00390625f;
    return fminf(fmaxf(r, -1.0f), 1.0f);
}

union V16 { v16h v; v8h h[2]; };

// ---------------------------------------------------------------------------
// Fallback path (no workspace / no async builtins): fused quantize + WMMA GEMM
// ---------------------------------------------------------------------------
__global__ __launch_bounds__(256) void conv_fixed_wmma(
        const float* __restrict__ x, const float* __restrict__ w,
        const float* __restrict__ bias, float* __restrict__ out)
{
    __shared__ alignas(32) _Float16 lA[2][64][32];
    __shared__ alignas(32) _Float16 lB[2][8][32][16];

    const int t     = threadIdx.x;
    const int lane  = t & 31;
    const int wv    = t >> 5;
    const int p0    = blockIdx.x * 64;
    const int nBase = blockIdx.y * 128;

    const int aRow = t >> 2;
    const int aCol = (t & 3) * 8;
    const int pa   = p0 + aRow;
    const int nImg = pa / PIX;
    const int rem  = pa - nImg * PIX;
    const int oh   = rem / OWW;
    const int ow   = rem - oh * OWW;
    const float* aBase = x + (((nImg*HH + oh)*WW) + ow)*CC + aCol;

    const int nn    = t & 127;
    const int khalf = t >> 7;
    const int tileS = nn >> 4;
    const int laneS = (nn & 15) + 16*khalf;
    const float* wBase = w + nBase + nn;

    float aR[8];
    float wR[16];
    {
        float4 a0 = *(const float4*)(aBase);
        float4 a1 = *(const float4*)(aBase + 4);
        aR[0]=a0.x; aR[1]=a0.y; aR[2]=a0.z; aR[3]=a0.w;
        aR[4]=a1.x; aR[5]=a1.y; aR[6]=a1.z; aR[7]=a1.w;
        const float* wp = wBase + (khalf*16)*FF;
        #pragma unroll
        for (int j = 0; j < 16; ++j) wR[j] = wp[j*FF];
    }

    v8f acc[4] = {};
    const int mW    = wv & 3;
    const int nH    = wv >> 2;
    const int aRowF = mW*16 + (lane & 15);
    const int cb    = (lane >> 4) * 16;

    for (int kb = 0; kb < KSTEPS; ++kb) {
        const int buf = kb & 1;
        {
            v8h qa;
            #pragma unroll
            for (int i = 0; i < 8; ++i) qa[i] = (_Float16)qf(aR[i]);
            *(v8h*)&lA[buf][aRow][aCol] = qa;
            v8h q0, q1;
            #pragma unroll
            for (int j = 0; j < 8; ++j) q0[j] = (_Float16)qf(wR[j]);
            #pragma unroll
            for (int j = 0; j < 8; ++j) q1[j] = (_Float16)qf(wR[8+j]);
            _Float16* bp = &lB[buf][tileS][laneS][0];
            *(v8h*)(bp)     = q0;
            *(v8h*)(bp + 8) = q1;
        }
        if (kb + 1 < KSTEPS) {
            const int kn = kb + 1;
            const int kh = kn / 12;
            const int r2 = kn - kh*12;
            const int kw = r2 >> 2;
            const int c0 = (r2 & 3) << 5;
            const float* ap = aBase + (kh*WW + kw)*CC + c0;
            float4 a0 = *(const float4*)(ap);
            float4 a1 = *(const float4*)(ap + 4);
            aR[0]=a0.x; aR[1]=a0.y; aR[2]=a0.z; aR[3]=a0.w;
            aR[4]=a1.x; aR[5]=a1.y; aR[6]=a1.z; aR[7]=a1.w;
            const float* wp = wBase + (((kh*3 + kw)*CC) + c0 + khalf*16)*FF;
            #pragma unroll
            for (int j = 0; j < 16; ++j) wR[j] = wp[j*FF];
        }
        __syncthreads();
        V16 af;
        af.h[0] = *(const v8h*)&lA[buf][aRowF][cb];
        af.h[1] = *(const v8h*)&lA[buf][aRowF][cb + 8];
        #pragma unroll
        for (int i = 0; i < 4; ++i) {
            V16 bf;
            const _Float16* bp = &lB[buf][nH*4 + i][lane][0];
            bf.h[0] = *(const v8h*)(bp);
            bf.h[1] = *(const v8h*)(bp + 8);
            acc[i] = __builtin_amdgcn_wmma_f32_16x16x32_f16(
                false, af.v, false, bf.v, (short)0, acc[i], false, false);
        }
    }

    const int nIdx = lane & 15;
    const int mHi  = (lane >> 4) * 8;
    #pragma unroll
    for (int i = 0; i < 4; ++i) {
        const int f = nBase + nH*64 + i*16 + nIdx;
        #pragma unroll
        for (int r = 0; r < 8; ++r) {
            const int pOut = p0 + mW*16 + mHi + r;
            const int rp   = pOut - (pOut / PIX) * PIX;
            out[pOut*FF + f] = fmaxf(qf(acc[i][r]) + bias[rp*FF + f], 0.0f);
        }
    }
}

#if HAVE_ASYNC
// ---------------------------------------------------------------------------
// Fast path: pre-quantize to f16 workspace, then async-to-LDS staged WMMA GEMM
// ---------------------------------------------------------------------------
// Builtin expects pointers to a 16-byte int vector (clang "V4i"), AS1 / AS3.
typedef int v4i __attribute__((vector_size(16)));
typedef __attribute__((address_space(1))) v4i* gp128_t;
typedef __attribute__((address_space(3))) v4i* lp128_t;

__device__ __forceinline__ void asyncCp16(const void* g, void* l) {
    __builtin_amdgcn_global_load_async_to_lds_b128((gp128_t)g, (lp128_t)l, 0, 0);
}
__device__ __forceinline__ void asyncCp16off16(const void* g, void* l) {
    // single immediate offset applies to BOTH global and LDS addresses (ISA 10.x)
    __builtin_amdgcn_global_load_async_to_lds_b128((gp128_t)g, (lp128_t)l, 16, 0);
}
__device__ __forceinline__ void waitAsync0() {
#if __has_builtin(__builtin_amdgcn_s_wait_asynccnt)
    __builtin_amdgcn_s_wait_asynccnt(0);
#else
    asm volatile("s_wait_asynccnt 0x0" ::: "memory");
#endif
}

__global__ __launch_bounds__(256) void quant_x_f16(
        const float* __restrict__ x, _Float16* __restrict__ xq)
{
    const int i = (blockIdx.x * 256 + threadIdx.x) * 4;
    float4 v = *(const float4*)(x + i);
    v4h q;
    q[0] = (_Float16)qf(v.x); q[1] = (_Float16)qf(v.y);
    q[2] = (_Float16)qf(v.z); q[3] = (_Float16)qf(v.w);
    *(v4h*)(xq + i) = q;
}

// wqT[f][k]: K-major per output channel so a B fragment is 32 contiguous bytes
__global__ __launch_bounds__(256) void quant_w_f16T(
        const float* __restrict__ w, _Float16* __restrict__ wqT)
{
    const int k = blockIdx.x;     // 0..1151  (k = (kh*3+kw)*128 + c)
    const int f = threadIdx.x;    // 0..255
    wqT[f*KTOT + k] = (_Float16)qf(w[k*FF + f]);
}

__global__ __launch_bounds__(256) void conv_wmma_async(
        const _Float16* __restrict__ xq, const _Float16* __restrict__ wqT,
        const float* __restrict__ bias, float* __restrict__ out)
{
    __shared__ alignas(32) _Float16 lA[2][64][32];        // 8 KB
    __shared__ alignas(32) _Float16 lB[2][16][32][16];    // 32 KB

    const int t    = threadIdx.x;
    const int lane = t & 31;
    const int wv   = t >> 5;
    const int p0   = blockIdx.x * 64;       // 64 M x 256 N per workgroup

    // ---- A staging: 1 async b128 per thread (64 rows x 4 chunks) ----
    const int aRow = t >> 2;
    const int c4   = (t & 3) * 8;
    const int pa   = p0 + aRow;
    const int nImg = pa / PIX;
    const int rem  = pa - nImg * PIX;
    const int oh   = rem / OWW;
    const int ow   = rem - oh * OWW;
    const _Float16* aG = xq + (((nImg*HH + oh)*WW) + ow)*CC + c4;

    // ---- B staging: 2 tiles x 2 async b128 per thread ----
    const int tb  = t >> 5;                 // tiles tb and tb+8
    const int lnS = t & 31;
    const int kIn = (lnS >> 4) * 16;        // contiguous-k B layout used here: lanes 16-31 hold K 16..31
    const _Float16* bG0 = wqT + ((tb   *16) + (lnS & 15))*KTOT + kIn;
    const _Float16* bG1 = wqT + (((tb+8)*16) + (lnS & 15))*KTOT + kIn;

    v8f acc[8] = {};
    const int mW    = wv & 3;
    const int nH    = wv >> 2;              // wave: 16 M x 128 N (8 tiles)
    const int aRowF = mW*16 + (lane & 15);
    const int cb    = (lane >> 4) * 16;

    // prologue: stage K-block 0 into buffer 0
    {
        asyncCp16(aG, &lA[0][aRow][c4]);
        asyncCp16(bG0, &lB[0][tb][lnS][0]);
        asyncCp16off16(bG0, &lB[0][tb][lnS][0]);
        asyncCp16(bG1, &lB[0][tb+8][lnS][0]);
        asyncCp16off16(bG1, &lB[0][tb+8][lnS][0]);
    }

    for (int kb = 0; kb < KSTEPS; ++kb) {
        const int buf = kb & 1;
        waitAsync0();        // my batch kb landed in LDS
        __syncthreads();     // everyone's batch kb landed; prev reads fenced

        if (kb + 1 < KSTEPS) {   // stage next block, overlapped with WMMAs
            const int kn = kb + 1;
            const int kh = kn / 12;
            const int r2 = kn - kh*12;
            const int kw = r2 >> 2;
            const int c0 = (r2 & 3) << 5;
            const int nb = buf ^ 1;
            asyncCp16(aG + (kh*WW + kw)*CC + c0, &lA[nb][aRow][c4]);
            const _Float16* bg0 = bG0 + kn*32;
            asyncCp16(bg0, &lB[nb][tb][lnS][0]);
            asyncCp16off16(bg0, &lB[nb][tb][lnS][0]);
            const _Float16* bg1 = bG1 + kn*32;
            asyncCp16(bg1, &lB[nb][tb+8][lnS][0]);
            asyncCp16off16(bg1, &lB[nb][tb+8][lnS][0]);
        }

        V16 af;
        af.h[0] = *(const v8h*)&lA[buf][aRowF][cb];
        af.h[1] = *(const v8h*)&lA[buf][aRowF][cb + 8];

        #pragma unroll
        for (int g = 0; g < 2; ++g) {
            V16 bf[4];
            #pragma unroll
            for (int i = 0; i < 4; ++i) {
                const _Float16* bp = &lB[buf][nH*8 + g*4 + i][lane][0];
                bf[i].h[0] = *(const v8h*)(bp);
                bf[i].h[1] = *(const v8h*)(bp + 8);
            }
            #pragma unroll
            for (int i = 0; i < 4; ++i) {
                acc[g*4 + i] = __builtin_amdgcn_wmma_f32_16x16x32_f16(
                    false, af.v, false, bf[i].v, (short)0, acc[g*4 + i],
                    false, false);
            }
        }
    }

    // epilogue: quantize + bias + relu
    const int nIdx = lane & 15;
    const int mHi  = (lane >> 4) * 8;
    #pragma unroll
    for (int i = 0; i < 8; ++i) {
        const int f = (nH*8 + i)*16 + nIdx;
        #pragma unroll
        for (int r = 0; r < 8; ++r) {
            const int pOut = p0 + mW*16 + mHi + r;
            const int rp   = pOut - (pOut / PIX) * PIX;
            out[pOut*FF + f] = fmaxf(qf(acc[i][r]) + bias[rp*FF + f], 0.0f);
        }
    }
}
#endif // HAVE_ASYNC

extern "C" void kernel_launch(void* const* d_in, const int* in_sizes, int n_in,
                              void* d_out, int out_size, void* d_ws, size_t ws_size,
                              hipStream_t stream) {
    const float* x  = (const float*)d_in[0];
    const float* w  = (const float*)d_in[1];
    const float* b  = (const float*)d_in[2];
    float* out      = (float*)d_out;
    (void)in_sizes; (void)n_in; (void)out_size;

#if HAVE_ASYNC
    if (ws_size >= WS_NEEDED && d_ws != nullptr) {
        _Float16* xq  = (_Float16*)d_ws;
        _Float16* wqT = (_Float16*)((char*)d_ws + XQ_BYTES);
        quant_x_f16<<<XTOT/4/256, 256, 0, stream>>>(x, xq);
        quant_w_f16T<<<KTOT, 256, 0, stream>>>(w, wqT);
        conv_wmma_async<<<MTOT/64, 256, 0, stream>>>(xq, wqT, b, out);
        return;
    }
#endif
    (void)d_ws; (void)ws_size;
    dim3 grid(MTOT/64, 2);
    conv_fixed_wmma<<<grid, 256, 0, stream>>>(x, w, b, out);
}
